// Net_en_65609920413744
// MI455X (gfx1250) — compile-verified, matrix operand order checked by reading, not run on hardware
//
#include <hip/hip_runtime.h>
#include <math.h>

#define N_NODES 50000
#define E_EDGES 800000
#define DIM 128
#define L_LAYERS 3
#define G_GRAPHS 1000
#define NEG_SLOPE 0.2f
#define MT_PER_BLOCK 16          // 16-row tiles handled per block (256 rows)

typedef __attribute__((ext_vector_type(16))) _Float16 v16h;
typedef __attribute__((ext_vector_type(8)))  float    v8f;

// ---------------------------------------------------------------- utilities

__global__ void k_fill(float* __restrict__ p, float v, int n) {
    int i = blockIdx.x * blockDim.x + threadIdx.x;
    if (i < n) p[i] = v;
}

__device__ inline void atomicMaxF(float* addr, float val) {
    int* ai = (int*)addr;
    int old = __float_as_int(*addr);
    while (__int_as_float(old) < val) {
        int assumed = old;
        old = atomicCAS(ai, assumed, __float_as_int(val));
        if (old == assumed) break;
    }
}

// ---------------------------------------------------------------- WMMA GEMM
// C[M,128] = A[M,128] @ W[128,128]   (f32 in/out, f16 WMMA math)
// Block = 256 threads (8 waves). W is staged once into LDS pre-packed in the
// exact per-lane B-fragment layout; wave w owns column tile w; the block walks
// MT_PER_BLOCK row tiles. A loads are 4x float4 per K-block (no divergence:
// row index is clamped, stores are guarded).
__global__ void __launch_bounds__(256)
k_gemm_wmma(const float* __restrict__ A, const float* __restrict__ W,
            float* __restrict__ C, int M) {
    // [tile_n][kb][lane][t] : 8*4*32*16 halves = 32 KB
    __shared__ _Float16 wl[8 * 4 * 32 * 16];

    const int tid = threadIdx.x;

    // ---- stage W (f32 -> f16) into B-fragment layout ----
    for (int idx = tid; idx < DIM * DIM; idx += 256) {
        const int k = idx >> 7, col = idx & 127;
        const int tile_n = col >> 4, lane16 = col & 15;
        const int kb = k >> 5, kk = k & 31;
        const int hsel = (kk >> 3) & 1;                 // K-half select
        const int t = (kk & 7) + ((kk & 16) ? 8 : 0);   // fragment element
        wl[(((tile_n * 4 + kb) * 32) + (lane16 + 16 * hsel)) * 16 + t] =
            (_Float16)W[idx];
    }
    __syncthreads();

    const int wave   = tid >> 5;     // column tile owned by this wave (0..7)
    const int lane   = tid & 31;
    const int hsel   = lane >> 4;
    const int lane16 = lane & 15;
    const int col    = wave * 16 + lane16;
    const _Float16* bbase = &wl[(size_t)(wave * 4 * 32 + lane) * 16];

    for (int mt = 0; mt < MT_PER_BLOCK; ++mt) {
        const int tile_m = blockIdx.x * MT_PER_BLOCK + mt;
        const int row0 = tile_m * 16;
        if (row0 >= M) break;
        const int row = min(row0 + lane16, M - 1);      // clamp: no divergence
        const float* Ar = A + (size_t)row * DIM;

        v8f acc = {};
        #pragma unroll
        for (int kb = 0; kb < 4; ++kb) {
            const int K0 = kb * 32 + hsel * 8;
            // two runs of 8 consecutive K values -> 4x b128 loads
            const float4 r0 = *(const float4*)(Ar + K0);
            const float4 r1 = *(const float4*)(Ar + K0 + 4);
            const float4 r2 = *(const float4*)(Ar + K0 + 16);
            const float4 r3 = *(const float4*)(Ar + K0 + 20);
            v16h a;
            a[0]  = (_Float16)r0.x; a[1]  = (_Float16)r0.y;
            a[2]  = (_Float16)r0.z; a[3]  = (_Float16)r0.w;
            a[4]  = (_Float16)r1.x; a[5]  = (_Float16)r1.y;
            a[6]  = (_Float16)r1.z; a[7]  = (_Float16)r1.w;
            a[8]  = (_Float16)r2.x; a[9]  = (_Float16)r2.y;
            a[10] = (_Float16)r2.z; a[11] = (_Float16)r2.w;
            a[12] = (_Float16)r3.x; a[13] = (_Float16)r3.y;
            a[14] = (_Float16)r3.z; a[15] = (_Float16)r3.w;

            const v16h b = *(const v16h*)(bbase + (size_t)kb * 32 * 16);
            acc = __builtin_amdgcn_wmma_f32_16x16x32_f16(
                false, a, false, b, (short)0, acc, false, false);
        }
        // C/D layout: N = lane16, M = r + 8*hsel
        #pragma unroll
        for (int r = 0; r < 8; ++r) {
            const int orow = row0 + r + 8 * hsel;
            if (orow < M) C[(size_t)orow * DIM + col] = acc[r];
        }
    }
}

// ---------------------------------------------------------------- per-node scalars
__global__ void k_node_scalars(const float* __restrict__ h,
                               const float* __restrict__ a_src,
                               const float* __restrict__ a_dst,
                               float* __restrict__ hs, float* __restrict__ hd,
                               int Nn) {
    const int wave = (blockIdx.x * blockDim.x + threadIdx.x) >> 5;
    const int lane = threadIdx.x & 31;
    if (wave >= Nn) return;
    const float* hr = h + (size_t)wave * DIM;
    float s = 0.f, d = 0.f;
    #pragma unroll
    for (int t = 0; t < 4; ++t) {
        const int f = lane + 32 * t;
        const float v = hr[f];
        s += v * a_src[f];
        d += v * a_dst[f];
    }
    #pragma unroll
    for (int off = 16; off; off >>= 1) {
        s += __shfl_down(s, off, 32);
        d += __shfl_down(d, off, 32);
    }
    if (lane == 0) { hs[wave] = s; hd[wave] = d; }
}

// ---------------------------------------------------------------- edge passes
__device__ inline void edge_sd(const int* __restrict__ ei, int e, int E, int* s, int* d) {
    if (e < E) { *s = ei[e]; *d = ei[E + e]; }
    else       { *s = *d = e - E; }          // self loops appended
}

__global__ void k_edge_max(const int* __restrict__ ei,
                           const float* __restrict__ hs,
                           const float* __restrict__ hd,
                           float* __restrict__ m, int E, int Nn) {
    const int e = blockIdx.x * blockDim.x + threadIdx.x;
    if (e >= E + Nn) return;
    int s, d; edge_sd(ei, e, E, &s, &d);
    float v = hs[s] + hd[d];
    v = (v > 0.f) ? v : NEG_SLOPE * v;       // leaky relu
    atomicMaxF(&m[d], v);
}

__global__ void k_edge_expsum(const int* __restrict__ ei,
                              const float* __restrict__ hs,
                              const float* __restrict__ hd,
                              const float* __restrict__ m,
                              float* __restrict__ wbuf,
                              float* __restrict__ denom, int E, int Nn) {
    const int e = blockIdx.x * blockDim.x + threadIdx.x;
    if (e >= E + Nn) return;
    int s, d; edge_sd(ei, e, E, &s, &d);
    float v = hs[s] + hd[d];
    v = (v > 0.f) ? v : NEG_SLOPE * v;
    const float w = __expf(v - m[d]);
    wbuf[e] = w;
    atomicAdd(&denom[d], w);
}

// out[dst,:] += (w/denom[dst]) * h[src,:]  — one wave per edge, 4 feats/lane.
__global__ void k_edge_agg(const int* __restrict__ ei,
                           const float* __restrict__ wbuf,
                           const float* __restrict__ denom,
                           const float* __restrict__ h,
                           float* __restrict__ out, int E, int Nn) {
    const int e    = (blockIdx.x * blockDim.x + threadIdx.x) >> 5;
    const int lane = threadIdx.x & 31;
    if (e >= E + Nn) return;
    int s, d; edge_sd(ei, e, E, &s, &d);
    const float* hr = h + (size_t)s * DIM;
    __builtin_prefetch(hr + lane * 4, 0, 0);            // global_prefetch_b8
    const float alpha = wbuf[e] / denom[d];
    float* o = out + (size_t)d * DIM;
    #pragma unroll
    for (int t = 0; t < 4; ++t) {
        const int f = lane + 32 * t;
        atomicAdd(&o[f], hr[f] * alpha);
    }
}

// ---------------------------------------------------------------- elementwise
__global__ void k_bias_relu(float* __restrict__ buf, const float* __restrict__ b, int total) {
    const int i = blockIdx.x * blockDim.x + threadIdx.x;
    if (i >= total) return;
    const float v = buf[i] + b[i & (DIM - 1)];
    buf[i] = (v > 0.f) ? v : 0.f;
}

__global__ void k_pool(const float* __restrict__ act, const int* __restrict__ batchs,
                       float* __restrict__ g, int Nn) {
    const int i = blockIdx.x * blockDim.x + threadIdx.x;
    if (i >= Nn * DIM) return;
    const int n = i >> 7, f = i & (DIM - 1);
    atomicAdd(&g[(size_t)batchs[n] * DIM + f], act[i]);
}

// z = relu(z_pre + energy[g]*W_row128 + b) + y   (residual)
__global__ void k_lin1_post(float* __restrict__ z, const float* __restrict__ y,
                            const float* __restrict__ energy,
                            const float* __restrict__ w128,
                            const float* __restrict__ b, int Gn) {
    const int i = blockIdx.x * blockDim.x + threadIdx.x;
    if (i >= Gn * DIM) return;
    const int g = i >> 7, c = i & (DIM - 1);
    float v = z[i] + energy[g] * w128[c] + b[c];
    v = (v > 0.f) ? v : 0.f;
    z[i] = v + y[i];
}

// logits = z @ W3[128,2] + b3 ; log_softmax. One wave per graph.
__global__ void k_head(const float* __restrict__ z, const float* __restrict__ W3,
                       const float* __restrict__ b3, float* __restrict__ out, int Gn) {
    const int g    = (blockIdx.x * blockDim.x + threadIdx.x) >> 5;
    const int lane = threadIdx.x & 31;
    if (g >= Gn) return;
    const float* zr = z + (size_t)g * DIM;
    float p0 = 0.f, p1 = 0.f;
    #pragma unroll
    for (int t = 0; t < 4; ++t) {
        const int f = lane + 32 * t;
        const float v = zr[f];
        p0 += v * W3[f * 2 + 0];
        p1 += v * W3[f * 2 + 1];
    }
    #pragma unroll
    for (int off = 16; off; off >>= 1) {
        p0 += __shfl_down(p0, off, 32);
        p1 += __shfl_down(p1, off, 32);
    }
    if (lane == 0) {
        const float l0 = p0 + b3[0], l1 = p1 + b3[1];
        const float mx  = fmaxf(l0, l1);
        const float lse = mx + __logf(__expf(l0 - mx) + __expf(l1 - mx));
        out[g * 2 + 0] = l0 - lse;
        out[g * 2 + 1] = l1 - lse;
    }
}

// ---------------------------------------------------------------- launch

extern "C" void kernel_launch(void* const* d_in, const int* in_sizes, int n_in,
                              void* d_out, int out_size, void* d_ws, size_t ws_size,
                              hipStream_t stream) {
    const float* x      = (const float*)d_in[0];
    const float* energy = (const float*)d_in[1];
    const float* W1     = (const float*)d_in[2];
    const float* as1    = (const float*)d_in[3];
    const float* ad1    = (const float*)d_in[4];
    const float* b1     = (const float*)d_in[5];
    const float* Ws     = (const float*)d_in[6];
    const float* ass    = (const float*)d_in[7];
    const float* ads    = (const float*)d_in[8];
    const float* bs     = (const float*)d_in[9];
    const float* lin0_W = (const float*)d_in[10];
    const float* lin0_b = (const float*)d_in[11];
    const float* lin1_W = (const float*)d_in[12];
    const float* lin1_b = (const float*)d_in[13];
    const float* lins_W = (const float*)d_in[14];
    const float* lins_b = (const float*)d_in[15];
    const float* lin3_W = (const float*)d_in[16];
    const float* lin3_b = (const float*)d_in[17];
    const int*   ei     = (const int*)d_in[18];
    const int*   batchs = (const int*)d_in[19];
    float* out = (float*)d_out;

    // ---- workspace layout (floats) ----
    float* wsf   = (float*)d_ws;
    size_t off   = 0;
    float* hG    = wsf + off; off += (size_t)N_NODES * DIM;  // GEMM result h
    float* act   = wsf + off; off += (size_t)N_NODES * DIM;  // aggregated activations
    float* hs    = wsf + off; off += N_NODES;
    float* hd    = wsf + off; off += N_NODES;
    float* m     = wsf + off; off += N_NODES;
    float* denom = wsf + off; off += N_NODES;
    float* wbuf  = wsf + off; off += (size_t)E_EDGES + N_NODES;
    float* gpool = wsf + off; off += (size_t)G_GRAPHS * DIM;
    float* ybuf  = wsf + off; off += (size_t)G_GRAPHS * DIM;
    float* zbuf  = wsf + off; off += (size_t)G_GRAPHS * DIM;
    float* z2buf = wsf + off; off += (size_t)G_GRAPHS * DIM;

    const int ETOT = E_EDGES + N_NODES;
    const int tilesN = (N_NODES + 15) / 16;                       // 3125
    const int tilesG = (G_GRAPHS + 15) / 16;                      // 63
    const int gemmBlocksN = (tilesN + MT_PER_BLOCK - 1) / MT_PER_BLOCK;  // 196
    const int gemmBlocksG = (tilesG + MT_PER_BLOCK - 1) / MT_PER_BLOCK;  // 4

    const float* layer_in = x;
    for (int layer = 0; layer < 1 + L_LAYERS; ++layer) {
        const float* W   = (layer == 0) ? W1  : Ws  + (size_t)(layer - 1) * DIM * DIM;
        const float* as_ = (layer == 0) ? as1 : ass + (size_t)(layer - 1) * DIM;
        const float* ad_ = (layer == 0) ? ad1 : ads + (size_t)(layer - 1) * DIM;
        const float* b_  = (layer == 0) ? b1  : bs  + (size_t)(layer - 1) * DIM;

        // h = layer_in @ W     (WMMA)
        k_gemm_wmma<<<gemmBlocksN, 256, 0, stream>>>(layer_in, W, hG, N_NODES);
        // attention scalars
        k_node_scalars<<<(N_NODES + 3) / 4, 128, 0, stream>>>(hG, as_, ad_, hs, hd, N_NODES);
        // segment max / exp-sum
        k_fill<<<(N_NODES + 255) / 256, 256, 0, stream>>>(m, -3.0e38f, N_NODES);
        k_fill<<<(N_NODES + 255) / 256, 256, 0, stream>>>(denom, 0.0f, N_NODES);
        k_edge_max<<<(ETOT + 255) / 256, 256, 0, stream>>>(ei, hs, hd, m, E_EDGES, N_NODES);
        k_edge_expsum<<<(ETOT + 255) / 256, 256, 0, stream>>>(ei, hs, hd, m, wbuf, denom, E_EDGES, N_NODES);
        // weighted scatter-aggregate into act
        k_fill<<<((size_t)N_NODES * DIM + 255) / 256, 256, 0, stream>>>(act, 0.0f, N_NODES * DIM);
        k_edge_agg<<<((size_t)ETOT * 32 + 255) / 256, 256, 0, stream>>>(ei, wbuf, denom, hG, act, E_EDGES, N_NODES);
        // bias + relu
        k_bias_relu<<<((size_t)N_NODES * DIM + 255) / 256, 256, 0, stream>>>(act, b_, N_NODES * DIM);
        layer_in = act;
    }

    // global add pool
    k_fill<<<((size_t)G_GRAPHS * DIM + 255) / 256, 256, 0, stream>>>(gpool, 0.0f, G_GRAPHS * DIM);
    k_pool<<<((size_t)N_NODES * DIM + 255) / 256, 256, 0, stream>>>(act, batchs, gpool, N_NODES);

    // FC head
    k_gemm_wmma<<<gemmBlocksG, 256, 0, stream>>>(gpool, lin0_W, ybuf, G_GRAPHS);
    k_bias_relu<<<((size_t)G_GRAPHS * DIM + 255) / 256, 256, 0, stream>>>(ybuf, lin0_b, G_GRAPHS * DIM);

    // lin1: [y | energy] @ W[129,128] = y@W[0:128] + energy*W[128]
    k_gemm_wmma<<<gemmBlocksG, 256, 0, stream>>>(ybuf, lin1_W, zbuf, G_GRAPHS);
    k_lin1_post<<<((size_t)G_GRAPHS * DIM + 255) / 256, 256, 0, stream>>>(
        zbuf, ybuf, energy, lin1_W + (size_t)DIM * DIM, lin1_b, G_GRAPHS);

    // lins[0]
    k_gemm_wmma<<<gemmBlocksG, 256, 0, stream>>>(zbuf, lins_W, z2buf, G_GRAPHS);
    k_bias_relu<<<((size_t)G_GRAPHS * DIM + 255) / 256, 256, 0, stream>>>(z2buf, lins_b, G_GRAPHS * DIM);

    // logits + log_softmax
    k_head<<<(G_GRAPHS + 3) / 4, 128, 0, stream>>>(z2buf, lin3_W, lin3_b, out, G_GRAPHS);
}
